// DynamicConvolution_22497038696611
// MI455X (gfx1250) — matrix operand order repaired and checked by reading, not compiled, not run on hardware
//
#include <hip/hip_runtime.h>
#include <hip/hip_bf16.h>
#include <stdint.h>

// Problem constants (from reference): B=4, T=1024, C=1024, H=16, K=31
#define BB   4
#define TT   1024
#define CC   1024
#define HH   16
#define KW   31
#define PADW 15
#define DH   64          // CC / HH
#define HK   496         // HH * KW
#define HKP  512         // padded N for GEMM2
#define MM   (BB * TT)   // 4096 rows

typedef __attribute__((ext_vector_type(16))) __bf16 v16bf;
typedef __attribute__((ext_vector_type(8)))  float  v8f;

#if __has_builtin(__builtin_amdgcn_global_load_async_to_lds_b128) && \
    __has_builtin(__builtin_amdgcn_s_wait_asynccnt)
#define ASYNC_LDS 1
#else
#define ASYNC_LDS 0
#endif

#if ASYNC_LDS
typedef int v4i_ __attribute__((vector_size(16)));
typedef __attribute__((address_space(1))) v4i_ gas_v4i;   // global (AS1)
typedef __attribute__((address_space(3))) v4i_ las_v4i;   // LDS (AS3)
#endif

static __device__ __forceinline__ unsigned short f32_to_bf16_rne(float f) {
    union { float f; unsigned u; } x; x.f = f;
    unsigned u = x.u;
    unsigned r = u + 0x7FFFu + ((u >> 16) & 1u);
    if ((u & 0x7F800000u) == 0x7F800000u) r = u;   // inf/nan: truncate
    return (unsigned short)(r >> 16);
}

// 16-byte global -> LDS mover: async on CDNA5 (bypasses VGPRs, ASYNCcnt),
// synchronous fallback otherwise.
static __device__ __forceinline__ void stage16(const uint32_t* __restrict__ g,
                                               uint32_t* l) {
#if ASYNC_LDS
    __builtin_amdgcn_global_load_async_to_lds_b128(
        (gas_v4i*)(uintptr_t)g,
        (las_v4i*)(uint32_t)(uintptr_t)l,
        0, 0);
#else
    uint4 v = *(const uint4*)g;
    l[0] = v.x; l[1] = v.y; l[2] = v.z; l[3] = v.w;
#endif
}

static __device__ __forceinline__ void stage_fence() {
#if ASYNC_LDS
    __builtin_amdgcn_s_wait_asynccnt(0);
#endif
}

// ---------------------------------------------------------------------------
// f32 -> bf16 convert (grid-stride)
// ---------------------------------------------------------------------------
__global__ void dynconv_cvt_bf16(const float* __restrict__ in,
                                 unsigned short* __restrict__ out, int n) {
    int i = blockIdx.x * blockDim.x + threadIdx.x;
    int stride = gridDim.x * blockDim.x;
    for (; i < n; i += stride) out[i] = f32_to_bf16_rne(in[i]);
}

// ---------------------------------------------------------------------------
// C[m,n] = sum_k A[m,k]*B[n,k] + bias[n]
// A: bf16 [M][Kd] row-major, B: bf16 [>=Nreal][Kd] row-major (W, so B^T),
// C: f32 [M][ldc].
// Block tile 128(M) x 64(N), BK=32 (one bf16 WMMA k-step).
// 256 threads = 8 waves in a 4x2 grid; each wave owns a 32x32 output tile
// (2 A-frags x 2 B-frags -> 4 v_wmma per k-step).
// Fragment gathers follow CDNA5 ISA 7.12.2 VGPR layouts.
// ---------------------------------------------------------------------------
__global__ __launch_bounds__(256)
void dynconv_gemm_bf16(const unsigned short* __restrict__ A,
                       const unsigned short* __restrict__ Bw,
                       const float* __restrict__ bias,
                       float* __restrict__ Cout,
                       int M, int Nreal, int Kd, int ldc) {
    __shared__ uint32_t As[128 * 16];  // 128 rows x 32 bf16 (16 dwords) = 8KB
    __shared__ uint32_t Bs[64 * 16];   // 64 N-rows x 32 bf16 along K   = 4KB

    const int tid  = threadIdx.x;
    const int lane = tid & 31;
    const int wave = tid >> 5;
    const int lo   = lane & 15;
    const int hi   = lane >> 4;

    const int m0 = blockIdx.y * 128;
    const int n0 = blockIdx.x * 64;
    const int wm = (wave >> 1) * 32;   // wave row offset (0,32,64,96)
    const int wn = (wave & 1) * 32;    // wave col offset (0,32)

    const uint32_t* Ag = (const uint32_t*)A;
    const uint32_t* Bg = (const uint32_t*)Bw;
    const int strideU = Kd >> 1;       // dwords per row

    // A staging: thread -> row tid>>1 (0..127), dword half (tid&1)*8, 2 x b128
    const int aRow = tid >> 1;
    const int aCol = (tid & 1) * 8;
    // B staging: thread -> row tid>>2 (0..63), dword col (tid&3)*4, 1 x b128
    const int bRow = tid >> 2;
    const int bCol = (tid & 3) * 4;
    const int gn    = n0 + bRow;
    const bool bOk  = (gn < Nreal);

    v8f acc00 = {}, acc01 = {}, acc10 = {}, acc11 = {};

    for (int k0 = 0; k0 < Kd; k0 += 32) {
        __syncthreads();   // previous iteration's LDS consumers done
        {
            const uint32_t* src = Ag + (size_t)(m0 + aRow) * strideU + (k0 >> 1) + aCol;
            stage16(src,     &As[aRow * 16 + aCol]);
            stage16(src + 4, &As[aRow * 16 + aCol + 4]);
        }
        if (bOk) {
            stage16(Bg + (size_t)gn * strideU + (k0 >> 1) + bCol,
                    &Bs[bRow * 16 + bCol]);
        } else {
            uint32_t* d = &Bs[bRow * 16 + bCol];
            d[0] = 0u; d[1] = 0u; d[2] = 0u; d[3] = 0u;
        }
        stage_fence();     // this wave's async copies complete
        __syncthreads();   // all waves' tiles visible

        // B fragments (32x16 K-major): lane column = wn+{0,16}+lo,
        // dwords: v<4 -> hi*4+v, v>=4 -> 8 + hi*4 + (v-4)
        union { uint32_t d[8]; v16bf v; } bf0, bf1;
        {
            const uint32_t* bp0 = &Bs[(wn + lo) * 16];
            const uint32_t* bp1 = &Bs[(wn + 16 + lo) * 16];
            #pragma unroll
            for (int vv = 0; vv < 8; ++vv) { int dw = (vv < 4) ? (hi * 4 + vv) : (8 + hi * 4 + vv - 4); bf0.d[vv] = bp0[dw]; bf1.d[vv] = bp1[dw]; }
        }
        // A fragments (16x32): row = wm+{0,16}+lo; dwords: v<4 -> hi*4+v,
        // v>=4 -> 8 + hi*4 + (v-4)
        union { uint32_t d[8]; v16bf v; } af0, af1;
        {
            const uint32_t* ap0 = &As[(wm + lo) * 16];
            const uint32_t* ap1 = &As[(wm + 16 + lo) * 16];
            #pragma unroll
            for (int vv = 0; vv < 4; ++vv) {
                af0.d[vv]     = ap0[hi * 4 + vv];
                af0.d[vv + 4] = ap0[8 + hi * 4 + vv];
                af1.d[vv]     = ap1[hi * 4 + vv];
                af1.d[vv + 4] = ap1[8 + hi * 4 + vv];
            }
        }

        acc00 = __builtin_amdgcn_wmma_f32_16x16x32_bf16(
                    false, af0.v, false, bf0.v, (short)0, acc00, false, false);
        acc01 = __builtin_amdgcn_wmma_f32_16x16x32_bf16(
                    false, af0.v, false, bf1.v, (short)0, acc01, false, false);
        acc10 = __builtin_amdgcn_wmma_f32_16x16x32_bf16(
                    false, af1.v, false, bf0.v, (short)0, acc10, false, false);
        acc11 = __builtin_amdgcn_wmma_f32_16x16x32_bf16(
                    false, af1.v, false, bf1.v, (short)0, acc11, false, false);
    }

    // epilogue: C/D layout -> element r: row = r + 8*hi, col = lo
    const int mBase = m0 + wm + hi * 8;
    int nA = n0 + wn + lo;
    int nBcol = n0 + wn + 16 + lo;
    float bvA = (nA    < Nreal && bias) ? bias[nA]    : 0.0f;
    float bvB = (nBcol < Nreal && bias) ? bias[nBcol] : 0.0f;
    #pragma unroll
    for (int r = 0; r < 8; ++r) {
        int mA = mBase + r;
        if (nA < Nreal) {
            Cout[(size_t)mA * ldc + nA]        = acc00[r] + bvA;
            Cout[(size_t)(mA + 16) * ldc + nA] = acc10[r] + bvA;
        }
        if (nBcol < Nreal) {
            Cout[(size_t)mA * ldc + nBcol]        = acc01[r] + bvB;
            Cout[(size_t)(mA + 16) * ldc + nBcol] = acc11[r] + bvB;
        }
    }
}

// ---------------------------------------------------------------------------
// GLU: x = a * sigmoid(g);  x1 is [rows][2C]; outputs f32 + bf16 copies
// ---------------------------------------------------------------------------
__global__ void dynconv_glu(const float* __restrict__ x1,
                            float* __restrict__ xf,
                            unsigned short* __restrict__ xb, int rows) {
    int i = blockIdx.x * blockDim.x + threadIdx.x;
    int total = rows * CC;
    int stride = gridDim.x * blockDim.x;
    for (; i < total; i += stride) {
        int r = i / CC, c = i % CC;
        float a = x1[(size_t)r * (2 * CC) + c];
        float g = x1[(size_t)r * (2 * CC) + CC + c];
        float v = a * (1.0f / (1.0f + __expf(-g)));
        xf[i] = v;
        xb[i] = f32_to_bf16_rne(v);
    }
}

// ---------------------------------------------------------------------------
// Banded softmax (+/-15 window) + dynamic conv + masking.
// One wave per (b,h,t); lane handles channels h*64+lane and h*64+lane+32.
// ---------------------------------------------------------------------------
__global__ __launch_bounds__(256)
void dynconv_softmax_conv(const float* __restrict__ wlog,      // [M][HKP]
                          const float* __restrict__ xf,        // [M][C]
                          const unsigned char* __restrict__ mask,   // [B][T]
                          const unsigned char* __restrict__ kpm,    // [B][T]
                          unsigned short* __restrict__ outb) { // [M][C]
    int wid  = blockIdx.x * 8 + (threadIdx.x >> 5);
    int lane = threadIdx.x & 31;
    if (wid >= BB * HH * TT) return;
    int t = wid % TT;
    int h = (wid / TT) % HH;
    int b = wid / (TT * HH);

    const float* wl = wlog + (size_t)(b * TT + t) * HKP + h * KW;
    int s_lo = (t - PADW < 0) ? 0 : t - PADW;
    int s_hi = (t + PADW >= TT) ? TT - 1 : t + PADW;

    float mx = -3.4e38f;
    for (int s = s_lo; s <= s_hi; ++s) mx = fmaxf(mx, wl[s - t + PADW]);
    float sum = 0.0f;
    for (int s = s_lo; s <= s_hi; ++s) sum += __expf(wl[s - t + PADW] - mx);
    float inv = 1.0f / sum;

    float acc0 = 0.0f, acc1 = 0.0f;
    int c0 = h * DH + lane;
    for (int s = s_lo; s <= s_hi; ++s) {
        float p = __expf(wl[s - t + PADW] - mx) * inv;
        const float* xr = xf + (size_t)(b * TT + s) * CC;
        acc0 += p * xr[c0];
        acc1 += p * xr[c0 + 32];
    }
    bool keep = mask[b * TT + t] && !kpm[b * TT + t];
    if (!keep) { acc0 = 0.0f; acc1 = 0.0f; }
    unsigned short* orow = outb + (size_t)(b * TT + t) * CC;
    orow[c0]      = f32_to_bf16_rne(acc0);
    orow[c0 + 32] = f32_to_bf16_rne(acc1);
}

// ---------------------------------------------------------------------------
extern "C" void kernel_launch(void* const* d_in, const int* in_sizes, int n_in,
                              void* d_out, int out_size, void* d_ws, size_t ws_size,
                              hipStream_t stream) {
    const float* query  = (const float*)d_in[0];
    // d_in[1]=key, d_in[2]=value : unused by reference
    const unsigned char* mask = (const unsigned char*)d_in[3];
    const unsigned char* kpm  = (const unsigned char*)d_in[4];
    const float* w1     = (const float*)d_in[5];
    const float* b1     = (const float*)d_in[6];
    const float* wW     = (const float*)d_in[7];
    const float* bW     = (const float*)d_in[8];
    const float* w2     = (const float*)d_in[9];
    const float* b2     = (const float*)d_in[10];
    float* out          = (float*)d_out;

    char* ws = (char*)d_ws;
    size_t off = 0;
    auto alloc = [&](size_t bytes) { char* p = ws + off; off += (bytes + 255) & ~(size_t)255; return p; };

    unsigned short* qb    = (unsigned short*)alloc((size_t)MM * CC * 2);        // query bf16
    unsigned short* w1b   = (unsigned short*)alloc((size_t)2 * CC * CC * 2);    // w1 bf16
    unsigned short* wWb   = (unsigned short*)alloc((size_t)HK * CC * 2);        // w_weight bf16
    unsigned short* w2b   = (unsigned short*)alloc((size_t)CC * CC * 2);        // w2 bf16
    float*          x1    = (float*)alloc((size_t)MM * 2 * CC * 4);             // pre-GLU
    float*          xf    = (float*)alloc((size_t)MM * CC * 4);                 // GLU out f32
    unsigned short* xb    = (unsigned short*)alloc((size_t)MM * CC * 2);        // GLU out bf16
    float*          wlog  = (float*)alloc((size_t)MM * HKP * 4);                // conv logits
    unsigned short* convb = (unsigned short*)alloc((size_t)MM * CC * 2);        // conv out bf16

    // 1) converts
    dynconv_cvt_bf16<<<1024, 256, 0, stream>>>(query, qb, MM * CC);
    dynconv_cvt_bf16<<<1024, 256, 0, stream>>>(w1, w1b, 2 * CC * CC);
    dynconv_cvt_bf16<<<512, 256, 0, stream>>>(wW, wWb, HK * CC);
    dynconv_cvt_bf16<<<512, 256, 0, stream>>>(w2, w2b, CC * CC);

    // 2) x1 = q @ w1^T + b1   (M=4096, N=2048, K=1024)
    dynconv_gemm_bf16<<<dim3(2 * CC / 64, MM / 128), 256, 0, stream>>>(
        qb, w1b, b1, x1, MM, 2 * CC, CC, 2 * CC);

    // 3) GLU
    dynconv_glu<<<1024, 256, 0, stream>>>(x1, xf, xb, MM);

    // 4) wlog = x @ wW^T + bW  (N=496 padded to 512)
    dynconv_gemm_bf16<<<dim3(HKP / 64, MM / 128), 256, 0, stream>>>(
        xb, wWb, bW, wlog, MM, HK, CC, HKP);

    // 5) banded softmax + dynamic conv + masks
    dynconv_softmax_conv<<<(BB * HH * TT) / 8, 256, 0, stream>>>(
        wlog, xf, mask, kpm, convb);

    // 6) out = conv @ w2^T + b2
    dynconv_gemm_bf16<<<dim3(CC / 64, MM / 128), 256, 0, stream>>>(
        convb, w2b, b2, out, MM, CC, CC, CC);
}
